// PatchAnomalyDetector_39633958207644
// MI455X (gfx1250) — matrix operand order repaired and checked by reading, not compiled, not run on hardware
//
#include <hip/hip_runtime.h>

typedef __attribute__((ext_vector_type(16))) __bf16 v16bf;
typedef __attribute__((ext_vector_type(8)))  float  v8f;

#define NPATCH_ROWS 65536   // 64 * 32 * 32
#define EMBED 256
#define NPROTO 4096
#define KCONV 768           // 3*16*16
#define LDA 772             // padded LDS stride (floats) for embed kernel
#define LDF 260             // padded LDS stride (floats) for sim kernel

__device__ __forceinline__ v8f wmma_bf16(v16bf a, v16bf b, v8f c) {
    return __builtin_amdgcn_wmma_f32_16x16x32_bf16(
        /*neg_a=*/false, a, /*neg_b=*/false, b,
        /*c_mod=*/(short)0, c, /*reuse_a=*/false, /*reuse_b=*/false);
}

// ---------------------------------------------------------------------------
// Prep: split conv_w [256 x 768] f32 into hi/lo bf16, packed in B-fragment
// order: frag[((kstep*256 + n)*2 + half)*16 + i] = w[n][kstep*32 + half*16 + i]
// ---------------------------------------------------------------------------
__global__ void prep_w_kernel(const float* __restrict__ w,
                              __bf16* __restrict__ whi, __bf16* __restrict__ wlo) {
    int row  = blockIdx.x * (blockDim.x >> 5) + (threadIdx.x >> 5);
    int lane = threadIdx.x & 31;
    if (row >= EMBED) return;
    for (int idx = lane; idx < KCONV; idx += 32) {
        int kstep = idx >> 5;
        int r     = idx & 31;
        int half  = r >> 4;
        int i     = r & 15;
        int K     = kstep * 32 + half * 16 + i;
        float v   = w[row * KCONV + K];
        __bf16 h  = (__bf16)v;
        __bf16 l  = (__bf16)(v - (float)h);
        int o = ((kstep * EMBED + row) * 2 + half) * 16 + i;
        whi[o] = h;
        wlo[o] = l;
    }
}

// ---------------------------------------------------------------------------
// Prep: L2-normalize prototypes [4096 x 256], split hi/lo bf16, B-frag order.
// ---------------------------------------------------------------------------
__global__ void prep_p_kernel(const float* __restrict__ p,
                              __bf16* __restrict__ phi, __bf16* __restrict__ plo) {
    int row  = blockIdx.x * (blockDim.x >> 5) + (threadIdx.x >> 5);
    int lane = threadIdx.x & 31;
    if (row >= NPROTO) return;
    float s = 0.0f;
    for (int k = lane; k < EMBED; k += 32) {
        float v = p[row * EMBED + k];
        s += v * v;
    }
    #pragma unroll
    for (int off = 16; off >= 1; off >>= 1) s += __shfl_xor(s, off, 32);
    float inv = 1.0f / fmaxf(sqrtf(s), 1e-12f);
    for (int idx = lane; idx < EMBED; idx += 32) {
        int kstep = idx >> 5;
        int r     = idx & 31;
        int half  = r >> 4;
        int i     = r & 15;
        int K     = kstep * 32 + half * 16 + i;
        float v   = p[row * EMBED + K] * inv;
        __bf16 h  = (__bf16)v;
        __bf16 l  = (__bf16)(v - (float)h);
        int o = ((kstep * NPROTO + row) * 2 + half) * 16 + i;
        phi[o] = h;
        plo[o] = l;
    }
}

// ---------------------------------------------------------------------------
// Patch-embed GEMM (split-bf16 WMMA) + bias + LayerNorm -> feats [65536 x 256]
// One block = 16 patches (one M-tile). 8 waves, each owns 2 N-tiles of 16.
// ---------------------------------------------------------------------------
__global__ __launch_bounds__(256) void embed_ln_kernel(
    const float* __restrict__ x,
    const __bf16* __restrict__ whi, const __bf16* __restrict__ wlo,
    const float* __restrict__ cb, const float* __restrict__ gamma,
    const float* __restrict__ beta, float* __restrict__ feats) {

    __shared__ float As[16 * LDA];
    __shared__ float rsum[16];
    __shared__ float rsq[16];

    int t    = threadIdx.x;
    int wave = t >> 5;
    int lane = t & 31;
    int mt   = blockIdx.x;

    // Gather 16 patches x 768 floats into LDS (float4, fully coalesced rows).
    for (int v = t; v < 3072; v += 256) {
        int m  = v / 192;          // patch within tile
        int q  = v % 192;
        int k  = q * 4;            // k index within 768
        int c  = k >> 8;
        int ky = (k >> 4) & 15;
        int kx = k & 15;
        int patch = mt * 16 + m;
        int b  = patch >> 10;
        int l  = patch & 1023;
        int hp = l >> 5;
        int wp = l & 31;
        size_t g = (((size_t)(b * 3 + c) * 512) + (size_t)(hp * 16 + ky)) * 512
                 + (size_t)(wp * 16 + kx);
        *(float4*)(As + m * LDA + k) = *(const float4*)(x + g);
    }
    if (t < 16) { rsum[t] = 0.0f; rsq[t] = 0.0f; }
    __syncthreads();

    int nhalf  = lane >> 4;       // lane-half selects K sub-range
    int ml     = lane & 15;       // A row (M) / B column (N) within tile
    int kbaseA = nhalf ? 16 : 0;

    v8f acc0 = {}; v8f acc1 = {};
    int n0 = (wave * 2 + 0) * 16 + ml;
    int n1 = (wave * 2 + 1) * 16 + ml;

    for (int ks = 0; ks < 24; ++ks) {
        // Build A hi/lo fragments from LDS (A layout: contiguous K = 16*half + i).
        const float* ap = As + ml * LDA + ks * 32 + kbaseA;
        float4 a0 = *(const float4*)(ap);
        float4 a1 = *(const float4*)(ap + 4);
        float4 a2 = *(const float4*)(ap + 8);
        float4 a3 = *(const float4*)(ap + 12);
        float av[16] = {a0.x, a0.y, a0.z, a0.w, a1.x, a1.y, a1.z, a1.w,
                        a2.x, a2.y, a2.z, a2.w, a3.x, a3.y, a3.z, a3.w};
        v16bf ahi, alo;
        #pragma unroll
        for (int i = 0; i < 16; ++i) {
            __bf16 h = (__bf16)av[i];
            ahi[i] = h;
            alo[i] = (__bf16)(av[i] - (float)h);
        }
        // B fragments, pre-packed: one 32B vector load each.
        v16bf bh0 = *(const v16bf*)(whi + ((ks * EMBED + n0) * 2 + nhalf) * 16);
        v16bf bl0 = *(const v16bf*)(wlo + ((ks * EMBED + n0) * 2 + nhalf) * 16);
        v16bf bh1 = *(const v16bf*)(whi + ((ks * EMBED + n1) * 2 + nhalf) * 16);
        v16bf bl1 = *(const v16bf*)(wlo + ((ks * EMBED + n1) * 2 + nhalf) * 16);

        acc0 = wmma_bf16(ahi, bh0, acc0);
        acc0 = wmma_bf16(ahi, bl0, acc0);
        acc0 = wmma_bf16(alo, bh0, acc0);
        acc1 = wmma_bf16(ahi, bh1, acc1);
        acc1 = wmma_bf16(ahi, bl1, acc1);
        acc1 = wmma_bf16(alo, bh1, acc1);
    }

    // Bias (per channel n)
    float cb0 = cb[n0], cb1 = cb[n1];
    #pragma unroll
    for (int j = 0; j < 8; ++j) { acc0[j] += cb0; acc1[j] += cb1; }

    // LayerNorm statistics: per-row partial sums across this wave's 32 channels
    #pragma unroll
    for (int j = 0; j < 8; ++j) {
        float s = acc0[j] + acc1[j];
        float q = acc0[j] * acc0[j] + acc1[j] * acc1[j];
        #pragma unroll
        for (int off = 1; off < 16; off <<= 1) {
            s += __shfl_xor(s, off, 32);
            q += __shfl_xor(q, off, 32);
        }
        if (ml == 0) {
            atomicAdd(&rsum[nhalf * 8 + j], s);
            atomicAdd(&rsq [nhalf * 8 + j], q);
        }
    }
    __syncthreads();

    float g0 = gamma[n0], be0 = beta[n0];
    float g1 = gamma[n1], be1 = beta[n1];
    size_t rowbase = (size_t)mt * 16;
    #pragma unroll
    for (int j = 0; j < 8; ++j) {
        int m      = nhalf * 8 + j;    // C-frag row: lanes0-15 -> j, lanes16-31 -> j+8
        float mu   = rsum[m] * (1.0f / 256.0f);
        float var  = rsq[m] * (1.0f / 256.0f) - mu * mu;
        float inv  = rsqrtf(var + 1e-5f);
        float f0   = (acc0[j] - mu) * inv * g0 + be0;
        float f1   = (acc1[j] - mu) * inv * g1 + be1;
        size_t r   = (rowbase + (size_t)m) * EMBED;
        feats[r + n0] = f0;
        feats[r + n1] = f1;
    }
}

// ---------------------------------------------------------------------------
// Cosine-sim GEMM (split-bf16 WMMA) fused with row-max -> anomaly = 1 - max.
// One block = 16 feature rows. A frags (8 ksteps, hi+lo) register-resident;
// 8 waves sweep 32 proto N-tiles each (256 total = 4096 protos).
// ---------------------------------------------------------------------------
__global__ __launch_bounds__(256) void sim_max_kernel(
    const float* __restrict__ feats,
    const __bf16* __restrict__ phi, const __bf16* __restrict__ plo,
    float* __restrict__ anom) {

    __shared__ float Fs[16 * LDF];
    __shared__ float nrm[16];
    __shared__ float pmax[8][16];

    int t    = threadIdx.x;
    int wave = t >> 5;
    int lane = t & 31;
    int mt   = blockIdx.x;

    // Stage 16 x 256 feats into LDS.
    for (int v = t; v < 1024; v += 256) {
        int m = v >> 6;
        int q = v & 63;
        float4 d = *(const float4*)(feats + ((size_t)(mt * 16 + m)) * EMBED + q * 4);
        *(float4*)(Fs + m * LDF + q * 4) = d;
    }
    if (t < 16) nrm[t] = 0.0f;
    __syncthreads();

    // Row sum-of-squares (16 threads per row, 16 values each).
    {
        int m = t >> 4, c = t & 15;
        const float* p = Fs + m * LDF + c * 16;
        float s = 0.0f;
        #pragma unroll
        for (int i = 0; i < 16; ++i) s += p[i] * p[i];
        atomicAdd(&nrm[m], s);
    }
    __syncthreads();

    int nhalf  = lane >> 4;
    int ml     = lane & 15;
    int kbaseA = nhalf * 16;
    float inv  = 1.0f / fmaxf(sqrtf(nrm[ml]), 1e-12f);

    // Build all 8 A-fragment pairs once (norm folded in), keep in registers.
    v16bf AH[8], AL[8];
    #pragma unroll
    for (int ks = 0; ks < 8; ++ks) {
        const float* ap = Fs + ml * LDF + ks * 32 + kbaseA;
        float4 a0 = *(const float4*)(ap);
        float4 a1 = *(const float4*)(ap + 4);
        float4 a2 = *(const float4*)(ap + 8);
        float4 a3 = *(const float4*)(ap + 12);
        float av[16] = {a0.x, a0.y, a0.z, a0.w, a1.x, a1.y, a1.z, a1.w,
                        a2.x, a2.y, a2.z, a2.w, a3.x, a3.y, a3.z, a3.w};
        #pragma unroll
        for (int i = 0; i < 16; ++i) {
            float v  = av[i] * inv;
            __bf16 h = (__bf16)v;
            AH[ks][i] = h;
            AL[ks][i] = (__bf16)(v - (float)h);
        }
    }

    float vmax[8];
    #pragma unroll
    for (int j = 0; j < 8; ++j) vmax[j] = -1e30f;

    for (int ti = 0; ti < 32; ++ti) {
        int nt = wave * 32 + ti;
        int n  = nt * 16 + ml;
        v8f acc = {};
        #pragma unroll
        for (int ks = 0; ks < 8; ++ks) {
            v16bf bh = *(const v16bf*)(phi + ((ks * NPROTO + n) * 2 + nhalf) * 16);
            v16bf bl = *(const v16bf*)(plo + ((ks * NPROTO + n) * 2 + nhalf) * 16);
            acc = wmma_bf16(AH[ks], bh, acc);
            acc = wmma_bf16(AH[ks], bl, acc);
            acc = wmma_bf16(AL[ks], bh, acc);
        }
        #pragma unroll
        for (int j = 0; j < 8; ++j) vmax[j] = fmaxf(vmax[j], acc[j]);
    }

    // Cross-lane then cross-wave max reduction.
    #pragma unroll
    for (int j = 0; j < 8; ++j) {
        float m = vmax[j];
        #pragma unroll
        for (int off = 1; off < 16; off <<= 1) m = fmaxf(m, __shfl_xor(m, off, 32));
        if (ml == 0) pmax[wave][nhalf * 8 + j] = m;
    }
    __syncthreads();
    if (t < 16) {
        float m = pmax[0][t];
        #pragma unroll
        for (int w = 1; w < 8; ++w) m = fmaxf(m, pmax[w][t]);
        anom[(size_t)mt * 16 + t] = 1.0f - m;
    }
}

// ---------------------------------------------------------------------------
extern "C" void kernel_launch(void* const* d_in, const int* in_sizes, int n_in,
                              void* d_out, int out_size, void* d_ws, size_t ws_size,
                              hipStream_t stream) {
    const float* x      = (const float*)d_in[0];   // [64,3,512,512]
    const float* conv_w = (const float*)d_in[1];   // [256,3,16,16]
    const float* conv_b = (const float*)d_in[2];   // [256]
    const float* gamma  = (const float*)d_in[3];   // [256]
    const float* beta   = (const float*)d_in[4];   // [256]
    const float* protos = (const float*)d_in[5];   // [4096,256]

    float* out   = (float*)d_out;
    float* anom  = out;                 // [65536]
    float* feats = out + NPATCH_ROWS;   // [65536, 256]

    char* ws = (char*)d_ws;
    __bf16* whi = (__bf16*)(ws);                 // 24*256*32 bf16 = 384 KB
    __bf16* wlo = (__bf16*)(ws + 393216);
    __bf16* phi = (__bf16*)(ws + 786432);        // 8*4096*32 bf16 = 2 MB
    __bf16* plo = (__bf16*)(ws + 2883584);

    prep_w_kernel<<<64,   128, 0, stream>>>(conv_w, whi, wlo);
    prep_p_kernel<<<1024, 128, 0, stream>>>(protos, phi, plo);
    embed_ln_kernel<<<4096, 256, 0, stream>>>(x, whi, wlo, conv_b, gamma, beta, feats);
    sim_max_kernel<<<4096, 256, 0, stream>>>(feats, phi, plo, anom);
}
